// AutoregressiveRNN_82351702933949
// MI455X (gfx1250) — compile-verified, matrix operand order checked by reading, not run on hardware
//
#include <hip/hip_runtime.h>
#include <hip/hip_bf16.h>
#include <math.h>

// ---------------------------------------------------------------------------
// Problem constants (from reference): N=1024, S=128 (T=127 steps), V=64,
// E=512, H=1024.
// ---------------------------------------------------------------------------
#define NB    1024
#define SEQ   128
#define TSTEP 127
#define VOC   64
#define EDIM  512
#define HDIM  1024

typedef __bf16 bf16_t;
typedef __attribute__((ext_vector_type(16))) __bf16 v16bf;
typedef __attribute__((ext_vector_type(8)))  float  v8f;

static __device__ __forceinline__ v8f wmma_bf16(v16bf a, v16bf b, v8f c) {
    // D = A(16x32 bf16) * B(32x16 bf16) + C(16x16 f32)
    return __builtin_amdgcn_wmma_f32_16x16x32_bf16(
        /*neg_a=*/false, a, /*neg_b=*/false, b,
        /*c_mod=*/(short)0, c, /*reuse_a=*/false, /*reuse_b=*/false);
}

static __device__ __forceinline__ float sigmoidf_fast(float x) {
    return 1.0f / (1.0f + __expf(-x));
}

// ---------------------------------------------------------------------------
// f32 -> bf16 weight/activation conversion
// ---------------------------------------------------------------------------
__global__ void k_f32_to_bf16(const float* __restrict__ src,
                              bf16_t* __restrict__ dst, int n) {
    int i = blockIdx.x * blockDim.x + threadIdx.x;
    if (i < n) dst[i] = (bf16_t)src[i];
}

// ---------------------------------------------------------------------------
// Embedding gather: inp[n][t][e] = bf16(emb[x[n][t]][e]), t in [0, T)
// one block per (n,t) row
// ---------------------------------------------------------------------------
__global__ void k_embed(const int* __restrict__ x,
                        const float* __restrict__ emb,
                        bf16_t* __restrict__ out) {
    int row = blockIdx.x;             // n*T + t
    int n = row / TSTEP, t = row % TSTEP;
    int tok = x[n * SEQ + t];
    const float* e = emb + (size_t)tok * EDIM;
    bf16_t* o = out + (size_t)row * EDIM;
    for (int i = threadIdx.x; i < EDIM; i += blockDim.x)
        o[i] = (bf16_t)e[i];
}

__global__ void k_zero(float* __restrict__ p, int n) {
    int i = blockIdx.x * blockDim.x + threadIdx.x;
    if (i < n) p[i] = 0.0f;
}

// ---------------------------------------------------------------------------
// One GRU timestep, fused:  gates = x_t @ W_ih^T  (+)  h_{t-1} @ W_hh^T,
// then sigmoid/tanh gating, h_t written to Hbuf[:, t, :].
//
// Blocking (per MI455X reasoning): block = 64 batch rows x 128 hidden cols,
// 8 waves; each wave owns ONE 16-wide N-subtile (its gate rows of W) and ALL
// four 16-row M-subtiles. Per 32-deep K-chunk: 3 B loads (r/z/n, reused
// across 4 M-subtiles) + 4 A loads feed 12 WMMAs -> ~halves VMEM-per-WMMA
// vs naive tiling. Accumulators: 16 x v8f = 128 VGPRs (no spill).
// Weights are bf16 and L2-resident (23 MB << 192 MB L2) across all steps.
// ---------------------------------------------------------------------------
__global__ __launch_bounds__(256) void k_gru_step(
    const bf16_t* __restrict__ Xin, int Kin,         // [N][T][Kin]
    bf16_t* __restrict__ Hbuf,                        // [N][T][H]
    const bf16_t* __restrict__ Wih,                   // [3H][Kin] row-major
    const bf16_t* __restrict__ Whh,                   // [3H][H]   row-major
    const float* __restrict__ bih,                    // [3H]
    const float* __restrict__ bhh,                    // [3H]
    int t)
{
    const int lane = threadIdx.x & 31;
    const int wave = threadIdx.x >> 5;      // 0..7 : N subtile owned by wave
    const int j0   = blockIdx.x * 128;      // hidden-unit tile base
    const int n0   = blockIdx.y * 64;       // batch tile base
    const int r    = lane & 15;             // row/col within fragment
    const int kk   = (lane >> 4) * 16;      // K half selected by lane group
    const int jrow = j0 + wave * 16 + r;    // B fragment row (= output unit)

    const v8f vz = {0.f,0.f,0.f,0.f,0.f,0.f,0.f,0.f};
    v8f c_r[4]  = {vz, vz, vz, vz};
    v8f c_z[4]  = {vz, vz, vz, vz};
    v8f c_xn[4] = {vz, vz, vz, vz};
    v8f c_hn[4] = {vz, vz, vz, vz};

    // ---- input GEMM: gx = x_t @ W_ih^T ------------------------------------
    {
        const bf16_t* Br = Wih + (size_t)(jrow)            * Kin + kk;
        const bf16_t* Bz = Wih + (size_t)(HDIM + jrow)     * Kin + kk;
        const bf16_t* Bn = Wih + (size_t)(2 * HDIM + jrow) * Kin + kk;
        const bf16_t* Ab = Xin + ((size_t)(n0 + r) * TSTEP + t) * Kin + kk;
        const size_t  mStride = (size_t)16 * TSTEP * Kin;
        for (int k0 = 0; k0 < Kin; k0 += 32) {
            v16bf br = *(const v16bf*)(Br + k0);
            v16bf bz = *(const v16bf*)(Bz + k0);
            v16bf bn = *(const v16bf*)(Bn + k0);
#pragma unroll
            for (int mi = 0; mi < 4; ++mi) {
                v16bf a = *(const v16bf*)(Ab + (size_t)mi * mStride + k0);
                c_r[mi]  = wmma_bf16(a, br, c_r[mi]);
                c_z[mi]  = wmma_bf16(a, bz, c_z[mi]);
                c_xn[mi] = wmma_bf16(a, bn, c_xn[mi]);
            }
        }
    }

    // ---- hidden GEMM: gh = h_{t-1} @ W_hh^T  (h_0 == 0 -> skip) -----------
    if (t > 0) {
        const bf16_t* Br = Whh + (size_t)(jrow)            * HDIM + kk;
        const bf16_t* Bz = Whh + (size_t)(HDIM + jrow)     * HDIM + kk;
        const bf16_t* Bn = Whh + (size_t)(2 * HDIM + jrow) * HDIM + kk;
        const bf16_t* Ab =
            Hbuf + ((size_t)(n0 + r) * TSTEP + (t - 1)) * HDIM + kk;
        const size_t mStride = (size_t)16 * TSTEP * HDIM;
        for (int k0 = 0; k0 < HDIM; k0 += 32) {
            v16bf br = *(const v16bf*)(Br + k0);
            v16bf bz = *(const v16bf*)(Bz + k0);
            v16bf bn = *(const v16bf*)(Bn + k0);
#pragma unroll
            for (int mi = 0; mi < 4; ++mi) {
                v16bf a = *(const v16bf*)(Ab + (size_t)mi * mStride + k0);
                c_r[mi]  = wmma_bf16(a, br, c_r[mi]);
                c_z[mi]  = wmma_bf16(a, bz, c_z[mi]);
                c_hn[mi] = wmma_bf16(a, bn, c_hn[mi]);
            }
        }
    }

    // ---- gating + writeback ----------------------------------------------
    const int j = j0 + wave * 16 + (lane & 15);   // hidden unit (fixed/lane)
    const float bir = bih[j],            bhr = bhh[j];
    const float biz = bih[HDIM + j],     bhz = bhh[HDIM + j];
    const float bin = bih[2 * HDIM + j], bhn = bhh[2 * HDIM + j];
#pragma unroll
    for (int mi = 0; mi < 4; ++mi) {
#pragma unroll
        for (int i = 0; i < 8; ++i) {
            const int m = n0 + mi * 16 + i + ((lane >> 4) << 3);  // batch idx
            float rg = sigmoidf_fast(c_r[mi][i] + bir + bhr);
            float zg = sigmoidf_fast(c_z[mi][i] + biz + bhz);
            float hp = 0.0f;
            float hn = bhn;
            if (t > 0) {
                hp  = (float)Hbuf[((size_t)m * TSTEP + (t - 1)) * HDIM + j];
                hn += c_hn[mi][i];
            }
            float ng = tanhf(c_xn[mi][i] + bin + rg * hn);
            float hv = (1.0f - zg) * ng + zg * hp;
            Hbuf[((size_t)m * TSTEP + t) * HDIM + j] = (bf16_t)hv;
        }
    }
}

// ---------------------------------------------------------------------------
// FC + log-softmax + per-molecule NLL accumulation.
// Each block: 64 consecutive (n,t) rows x all 64 vocab columns (K=1024 WMMA),
// logits -> LDS, row-wise log-softmax, write log_probs, atomicAdd token loss.
// ---------------------------------------------------------------------------
__global__ __launch_bounds__(256) void k_fc_loss(
    const bf16_t* __restrict__ H2,       // [N*T][H]
    const bf16_t* __restrict__ Wfc,      // [V][H]
    const float*  __restrict__ bfc,      // [V]
    const int*    __restrict__ x,        // [N][S]
    float* __restrict__ log_probs,       // [N*T][V]
    float* __restrict__ mol)             // [N] (pre-zeroed)
{
    __shared__ float sm[64][VOC + 8];

    const int lane = threadIdx.x & 31;
    const int wave = threadIdx.x >> 5;
    const int mi   = wave >> 1;
    const int ni0  = (wave & 1) * 2;
    const int r0   = blockIdx.x * 64;    // flat (n,t) row base
    const int r    = lane & 15;
    const int kk   = (lane >> 4) * 16;

    const v8f vz = {0.f,0.f,0.f,0.f,0.f,0.f,0.f,0.f};
    v8f acc[2] = {vz, vz};

    const bf16_t* Arow = H2 + (size_t)(r0 + mi * 16 + r) * HDIM;
    for (int k0 = 0; k0 < HDIM; k0 += 32) {
        v16bf a = *(const v16bf*)(Arow + k0 + kk);
#pragma unroll
        for (int s = 0; s < 2; ++s) {
            const bf16_t* b =
                Wfc + (size_t)((ni0 + s) * 16 + r) * HDIM + k0 + kk;
            acc[s] = wmma_bf16(a, *(const v16bf*)b, acc[s]);
        }
    }

#pragma unroll
    for (int s = 0; s < 2; ++s) {
        const int v  = (ni0 + s) * 16 + (lane & 15);
        const float bias = bfc[v];
#pragma unroll
        for (int i = 0; i < 8; ++i) {
            const int ml = mi * 16 + i + ((lane >> 4) << 3);
            sm[ml][v] = acc[s][i] + bias;
        }
    }
    __syncthreads();

    if (threadIdx.x < 64) {
        const int rl = threadIdx.x;
        const int R  = r0 + rl;
        const int n  = R / TSTEP;
        const int t  = R % TSTEP;
        float mx = -1e30f;
        for (int v = 0; v < VOC; ++v) mx = fmaxf(mx, sm[rl][v]);
        float se = 0.0f;
        for (int v = 0; v < VOC; ++v) se += __expf(sm[rl][v] - mx);
        const float lse = mx + __logf(se);
        float* op = log_probs + (size_t)R * VOC;
        for (int v = 0; v < VOC; ++v) op[v] = sm[rl][v] - lse;
        const int tgt = x[n * SEQ + t + 1];
        if (tgt != 0)  // IGNORE == 0
            atomicAdd(&mol[n], -(sm[rl][tgt] - lse));
    }
}

// ---------------------------------------------------------------------------
// lengths, normalized per-molecule loss, mean loss
// ---------------------------------------------------------------------------
__global__ __launch_bounds__(1024) void k_finalize(
    const int* __restrict__ x, float* __restrict__ mol,
    float* __restrict__ loss)
{
    __shared__ float red[1024];
    const int n = threadIdx.x;
    int len = 0;
    for (int s = 0; s < SEQ; ++s) len += (x[n * SEQ + s] != 0) ? 1 : 0;
    const float ml = mol[n];
    red[n] = ml;
    __syncthreads();
    for (int off = 512; off > 0; off >>= 1) {
        if (n < off) red[n] += red[n + off];
        __syncthreads();
    }
    mol[n] = ml / (float)(len > 0 ? len : 1);
    if (n == 0) loss[0] = red[0] / (float)NB;
}

// ---------------------------------------------------------------------------
// Host-side launcher
// ---------------------------------------------------------------------------
extern "C" void kernel_launch(void* const* d_in, const int* in_sizes, int n_in,
                              void* d_out, int out_size, void* d_ws,
                              size_t ws_size, hipStream_t stream) {
    (void)in_sizes; (void)n_in; (void)out_size; (void)ws_size;

    const int*   x    = (const int*)  d_in[0];
    const float* emb  = (const float*)d_in[1];
    const float* Wih0 = (const float*)d_in[2];
    const float* Whh0 = (const float*)d_in[3];
    const float* bih0 = (const float*)d_in[4];
    const float* bhh0 = (const float*)d_in[5];
    const float* Wih1 = (const float*)d_in[6];
    const float* Whh1 = (const float*)d_in[7];
    const float* bih1 = (const float*)d_in[8];
    const float* bhh1 = (const float*)d_in[9];
    const float* Wfc  = (const float*)d_in[10];
    const float* bfc  = (const float*)d_in[11];

    float* out       = (float*)d_out;
    float* log_probs = out;                             // N*T*V
    float* mol       = out + (size_t)NB * TSTEP * VOC;  // N
    float* loss      = mol + NB;                        // 1

    // -------- workspace carve-up (all bf16, 256B aligned) ------------------
    char*  ws  = (char*)d_ws;
    size_t off = 0;
    auto carve = [&](size_t elems) -> bf16_t* {
        bf16_t* p = (bf16_t*)(ws + off);
        off += (elems * sizeof(bf16_t) + 255) & ~(size_t)255;
        return p;
    };
    bf16_t* wih0 = carve((size_t)3 * HDIM * EDIM);
    bf16_t* whh0 = carve((size_t)3 * HDIM * HDIM);
    bf16_t* wih1 = carve((size_t)3 * HDIM * HDIM);
    bf16_t* whh1 = carve((size_t)3 * HDIM * HDIM);
    bf16_t* wfc  = carve((size_t)VOC * HDIM);
    bf16_t* inp  = carve((size_t)NB * TSTEP * EDIM);
    bf16_t* h1   = carve((size_t)NB * TSTEP * HDIM);
    bf16_t* h2   = carve((size_t)NB * TSTEP * HDIM);

    auto cvt = [&](const float* s, bf16_t* d, int n) {
        k_f32_to_bf16<<<(n + 255) / 256, 256, 0, stream>>>(s, d, n);
    };
    cvt(Wih0, wih0, 3 * HDIM * EDIM);
    cvt(Whh0, whh0, 3 * HDIM * HDIM);
    cvt(Wih1, wih1, 3 * HDIM * HDIM);
    cvt(Whh1, whh1, 3 * HDIM * HDIM);
    cvt(Wfc,  wfc,  VOC * HDIM);

    k_embed<<<NB * TSTEP, 256, 0, stream>>>(x, emb, inp);
    k_zero<<<(NB + 255) / 256, 256, 0, stream>>>(mol, NB);

    const dim3 gridStep(HDIM / 128, NB / 64);  // 8 x 16 workgroups per step
    for (int t = 0; t < TSTEP; ++t)
        k_gru_step<<<gridStep, 256, 0, stream>>>(inp, EDIM, h1, wih0, whh0,
                                                 bih0, bhh0, t);
    for (int t = 0; t < TSTEP; ++t)
        k_gru_step<<<gridStep, 256, 0, stream>>>(h1, HDIM, h2, wih1, whh1,
                                                 bih1, bhh1, t);

    k_fc_loss<<<(NB * TSTEP) / 64, 256, 0, stream>>>(h2, wfc, bfc, x,
                                                     log_probs, mol);
    k_finalize<<<1, 1024, 0, stream>>>(x, mol, loss);
}
